// CENPatchExpertPyTorch_31610959298739
// MI455X (gfx1250) — matrix operand order repaired and checked by reading, not compiled, not run on hardware
//
#include <hip/hip_runtime.h>
#include <cmath>

typedef _Float16 v2h  __attribute__((ext_vector_type(2)));
typedef _Float16 v8h  __attribute__((ext_vector_type(8)));
typedef _Float16 v16h __attribute__((ext_vector_type(16)));
typedef float    v8f  __attribute__((ext_vector_type(8)));

#define PATCH   11
#define HW      107
#define RH      97
#define NPATCH  9409      // 97*97
#define KPIX    121
#define MT      128       // patches per workgroup
#define NTILES  74        // ceil(9409/128)
#define FP      136       // f16 LDS row pitch (272B = 16B aligned, 4-bank rotate)
#define H2P     132       // f32 LDS row pitch for h2
#define XROWS   13        // max image rows a 128-patch tile touches (3 patch rows + 10)
#define XP      112       // image-stage row pitch (floats)
#define EPSV    1e-10f

__device__ __forceinline__ v16h cat16(v8h lo, v8h hi) {
    v16h r;
#pragma unroll
    for (int i = 0; i < 8; ++i) { r[i] = lo[i]; r[8 + i] = hi[i]; }
    return r;
}

// tanh via native exp/rcp: exact at +-inf, ~1e-6 abs error (below f16 noise floor)
__device__ __forceinline__ float fast_tanh(float v) {
    float e = __expf(2.0f * v);
    return 1.0f - 2.0f * __builtin_amdgcn_rcpf(e + 1.0f);
}
__device__ __forceinline__ float fast_sigmoid(float v) {
    return __builtin_amdgcn_rcpf(1.0f + __expf(-v));
}

__global__ __launch_bounds__(256) void cen_patch_expert_kernel(
    const float* __restrict__ x,  const float* __restrict__ W1, const float* __restrict__ b1,
    const float* __restrict__ W2, const float* __restrict__ b2,
    const float* __restrict__ W3, const float* __restrict__ b3,
    float* __restrict__ out)
{
    __shared__ __attribute__((aligned(16))) _Float16 featA[MT * FP];   // A for GEMM1 (patch features)
    __shared__ __attribute__((aligned(16))) _Float16 h1A  [MT * FP];   // A chunk for GEMM2
    __shared__ __attribute__((aligned(16))) _Float16 w1c  [128 * FP];  // W1^T chunk: [n_local][k]
    __shared__ __attribute__((aligned(16))) _Float16 w2c  [128 * FP];  // W2^T chunk: [n][k_local]
    __shared__ __attribute__((aligned(16))) float    h2s  [MT * H2P]; // layer-2 activations (f32)
    __shared__ __attribute__((aligned(16))) float    ximg [XROWS * XP]; // staged image rows (f32)

    const int tid  = threadIdx.x;
    const int wave = tid >> 5;
    const int lane = tid & 31;
    const int lmod = lane & 15;        // N (or M) index within a 16x16 tile
    const int lhalf = lane >> 4;       // lane half-group
    const int mrow = wave * 16;        // this wave's row stripe

    const int img  = blockIdx.x / NTILES;
    const int tile = blockIdx.x % NTILES;
    const int p0   = tile * MT;
    const int pr0  = p0 / RH;          // first patch row this tile touches

    // ---------------- Phase -1: stage the image rows this tile needs into LDS ----------------
    {
        const int plast = (p0 + MT - 1 < NPATCH - 1) ? p0 + MT - 1 : NPATCH - 1;
        const int nrows = (plast / RH) - pr0 + PATCH;         // <= 13
        const float* xb = x + (size_t)img * (HW * HW) + (size_t)pr0 * HW;
        for (int idx = tid; idx < nrows * HW; idx += 256) {
            const int r = idx / HW, c = idx - r * HW;
            ximg[r * XP + c] = xb[r * HW + c];
        }
    }
    __syncthreads();

    // ---------------- Phase 0: patch normalization -> featA ----------------
    if (tid < MT) {
        const int p = p0 + tid;
        _Float16* row = &featA[tid * FP];
        if (p < NPATCH) {
            const int pr = p / RH, pc = p % RH;
            const float* base = &ximg[(pr - pr0) * XP + pc];
            float s = 0.f, ss = 0.f;
            for (int i = 0; i < PATCH; ++i)
                for (int j = 0; j < PATCH; ++j) {
                    float v = base[i * XP + j];
                    s += v; ss += v * v;
                }
            const float mean = s * (1.0f / KPIX);
            float var = (ss - (float)KPIX * mean * mean) * (1.0f / (KPIX - 1));
            var = var < 0.f ? 0.f : var;
            const float inv = 1.0f / (sqrtf(var) + EPSV);
            row[0] = (_Float16)1.0f;
            for (int i = 0; i < PATCH; ++i)
                for (int j = 0; j < PATCH; ++j)
                    row[1 + i * PATCH + j] = (_Float16)((base[i * XP + j] - mean) * inv);
            for (int k = 122; k < 128; ++k) row[k] = (_Float16)0.f;
        } else {
            for (int k = 0; k < 128; ++k) row[k] = (_Float16)0.f;
        }
    }
    __syncthreads();

    // ---------------- Phases 1+2: chunked GEMM1 (122->512) fused into GEMM2 (512->128) ----------------
    v8f acc2[8] = {};   // layer-2 accumulators, live across chunks (8 N-tiles of 16)

#pragma unroll 1
    for (int kc = 0; kc < 4; ++kc) {
        // --- stage weight chunks to LDS as f16 (cooperative; weights are L2-resident) ---
        {
            const int r  = tid >> 1;
            const int k0 = (tid & 1) * 64;
            const float* w1row = W1 + (size_t)(kc * 128 + r) * 122;   // B1[k][n] = W1[n][k]
            const float* w2row = W2 + (size_t)r * 512 + kc * 128;     // B2[k][n] = W2[n][k]
#pragma unroll 4
            for (int k = k0; k < k0 + 64; k += 2) {
                v2h pk1, pk2;
                pk1[0] = (_Float16)((k     < 122) ? w1row[k]     : 0.f);
                pk1[1] = (_Float16)((k + 1 < 122) ? w1row[k + 1] : 0.f);
                pk2[0] = (_Float16)w2row[k];
                pk2[1] = (_Float16)w2row[k + 1];
                *(v2h*)&w1c[r * FP + k] = pk1;
                *(v2h*)&w2c[r * FP + k] = pk2;
            }
        }
        __syncthreads();

        // --- GEMM1: feat(128x128) x W1^T(128x128 chunk) -> acc1, per-wave 16-row stripe ---
        v8f acc1[8] = {};
#pragma unroll
        for (int kt = 0; kt < 4; ++kt) {
            const _Float16* ap = &featA[(mrow + lmod) * FP + kt * 32 + lhalf * 8];
            const v16h a = cat16(*(const v8h*)ap, *(const v8h*)(ap + 16));
            v16h bf[8];
#pragma unroll
            for (int nt = 0; nt < 8; ++nt) {     // preload all B fragments: 16 ds_loads in flight
                const _Float16* bp = &w1c[(nt * 16 + lmod) * FP + kt * 32 + lhalf * 8];
                bf[nt] = cat16(*(const v8h*)bp, *(const v8h*)(bp + 16));
            }
#pragma unroll
            for (int nt = 0; nt < 8; ++nt)
                acc1[nt] = __builtin_amdgcn_wmma_f32_16x16x32_f16(
                    false, a, false, bf[nt], (short)0, acc1[nt], false, false);
        }

        // --- bias + tanh -> h1A (f16), this wave's own 16 rows ---
#pragma unroll
        for (int nt = 0; nt < 8; ++nt) {
            const int col = nt * 16 + lmod;
            const float bias = b1[kc * 128 + col];
#pragma unroll
            for (int i = 0; i < 8; ++i) {
                const int rloc = mrow + lhalf * 8 + i;
                h1A[rloc * FP + col] = (_Float16)fast_tanh(acc1[nt][i] + bias);
            }
        }
        __syncthreads();

        // --- GEMM2 partial: h1A(16x128) x W2^T(128x128 chunk) accumulated into acc2 ---
#pragma unroll
        for (int kt = 0; kt < 4; ++kt) {
            const _Float16* ap = &h1A[(mrow + lmod) * FP + kt * 32 + lhalf * 8];
            const v16h a = cat16(*(const v8h*)ap, *(const v8h*)(ap + 16));
            v16h bf[8];
#pragma unroll
            for (int nt = 0; nt < 8; ++nt) {
                const _Float16* bp = &w2c[(nt * 16 + lmod) * FP + kt * 32 + lhalf * 8];
                bf[nt] = cat16(*(const v8h*)bp, *(const v8h*)(bp + 16));
            }
#pragma unroll
            for (int nt = 0; nt < 8; ++nt)
                acc2[nt] = __builtin_amdgcn_wmma_f32_16x16x32_f16(
                    false, a, false, bf[nt], (short)0, acc2[nt], false, false);
        }
        __syncthreads();   // before next chunk overwrites w1c/w2c
    }

    // ---------------- layer-2 bias + tanh -> h2s (f32) ----------------
#pragma unroll
    for (int nt = 0; nt < 8; ++nt) {
        const int col = nt * 16 + lmod;
        const float bias = b2[col];
#pragma unroll
        for (int i = 0; i < 8; ++i) {
            const int rloc = mrow + lhalf * 8 + i;
            h2s[rloc * H2P + col] = fast_tanh(acc2[nt][i] + bias);
        }
    }
    __syncthreads();

    // ---------------- layer 3: 128-dot + sigmoid, one thread per patch ----------------
    if (tid < MT) {
        const int p = p0 + tid;
        if (p < NPATCH) {
            float s = b3[0];
            const float* hrow = &h2s[tid * H2P];
#pragma unroll 8
            for (int n = 0; n < 128; ++n) s += hrow[n] * W3[n];
            out[(size_t)img * NPATCH + p] = fast_sigmoid(s);
        }
    }
}

extern "C" void kernel_launch(void* const* d_in, const int* in_sizes, int n_in,
                              void* d_out, int out_size, void* d_ws, size_t ws_size,
                              hipStream_t stream) {
    (void)in_sizes; (void)n_in; (void)d_ws; (void)ws_size; (void)out_size;
    const float* x  = (const float*)d_in[0];
    const float* W1 = (const float*)d_in[1];
    const float* b1 = (const float*)d_in[2];
    const float* W2 = (const float*)d_in[3];
    const float* b2 = (const float*)d_in[4];
    const float* W3 = (const float*)d_in[5];
    const float* b3 = (const float*)d_in[6];
    float* out = (float*)d_out;

    dim3 grid(32 * NTILES);   // 32 images x 74 patch-tiles
    dim3 block(256);          // 8 wave32 waves
    cen_patch_expert_kernel<<<grid, block, 0, stream>>>(x, W1, b1, W2, b2, W3, b3, out);
}
